// RNN_Model_23922967839315
// MI455X (gfx1250) — matrix operand order repaired and checked by reading, not compiled, not run on hardware
//
#include <hip/hip_runtime.h>

// ---------------------------------------------------------------------------
// 2-layer tanh RNN for MI455X (gfx1250, wave32, WMMA).
// All GEMMs are NT (X @ W^T, W row-major [out,in]) -> A and B WMMA fragments
// both load from row-major tiles, no transposes anywhere.
// bf16 inputs, f32 accumulate via v_wmma_f32_16x16x32_bf16.
// LDS staging uses CDNA5 async global->LDS copies (ASYNCcnt), double-buffered
// so the DMA for the next tile overlaps the WMMAs of the current tile.
// ---------------------------------------------------------------------------

typedef __attribute__((ext_vector_type(16))) __bf16 v16bf;
typedef __attribute__((ext_vector_type(8)))  float  v8f;

#define B_   64
#define T_   512
#define IN_  256
#define H_   512
#define OUT_ 256
#define BT_  (B_ * T_)
#define NWG  8          // workgroups in the recurrent scan (each owns H/NWG = 64 cols)

union FragU { v16bf v; uint4 u[2]; };

// ---- CDNA5 async global->LDS copy (GLOBAL_LOAD_ASYNC_TO_LDS_B128, ASYNCcnt).
// LDS destination operand: low 32 bits of the flat shared address are the LDS
// byte offset (ISA 10.2 aperture mapping). 16B alignment required on both ends.
__device__ __forceinline__ void async_copy_b128(const void* lds_dst, const void* gsrc) {
    const unsigned lds_off = (unsigned)(uintptr_t)lds_dst;
    const unsigned long long ga = (unsigned long long)(uintptr_t)gsrc;
    asm volatile("global_load_async_to_lds_b128 %0, %1, off"
                 :: "v"(lds_off), "v"(ga) : "memory");
}
__device__ __forceinline__ void wait_async0() {
    asm volatile("s_wait_asynccnt 0" ::: "memory");
}

// A fragment (16x32 bf16, M x K): lane m = lane&15, k-half h = lane>>4.
// Holds K in {8h..8h+7} and {16+8h..16+8h+7}: two 16B chunks.
__device__ __forceinline__ v16bf load_fragA(const __bf16* base, int stride, int lane) {
    const int row = lane & 15;
    const int kh  = (lane >> 4) << 3;          // 0 or 8
    const __bf16* p = base + row * stride + kh;
    FragU f;
    f.u[0] = *(const uint4*)(p);               // K = kh .. kh+7
    f.u[1] = *(const uint4*)(p + 16);          // K = 16+kh .. 16+kh+7
    return f.v;
}

// B fragment (32x16 bf16, K x N) built from B^T rows (= rows of W, row-major):
// lane n = lane&15 holds K in {8h..8h+7} and {16+8h..16+8h+7}: two 16B chunks.
__device__ __forceinline__ v16bf load_fragB(const __bf16* base, int stride, int lane) {
    const int row = lane & 15;
    const int ks  = (lane >> 4) << 3;          // 0 or 8
    const __bf16* p = base + row * stride + ks;
    FragU f;
    f.u[0] = *(const uint4*)(p);
    f.u[1] = *(const uint4*)(p + 16);
    return f.v;
}

// ---------------------------------------------------------------------------
// Generic NT GEMM: C[M,N] = A[M,K](bf16) * Bm[N,K](bf16)^T + bias[N]
// 256 threads = 8 waves, 64x64 C tile per WG. K staged in 64-wide LDS tiles,
// double-buffered: async copy of tile k+1 overlaps WMMAs on tile k.
// ---------------------------------------------------------------------------
__global__ __launch_bounds__(256) void gemm_nt(
    const __bf16* __restrict__ A, const __bf16* __restrict__ Bm,
    const float* __restrict__ bias, float* __restrict__ Cf,
    __bf16* __restrict__ Cbf, int M, int N, int K) {
    __shared__ __bf16 As[2][64][72];
    __shared__ __bf16 Bs[2][64][72];
    const int tid  = threadIdx.x;
    const int lane = tid & 31, wave = tid >> 5;
    const int wgM  = blockIdx.y * 64, wgN = blockIdx.x * 64;
    const int ntl   = wave & 3;          // local N tile 0..3
    const int mBase = (wave >> 2) * 32;  // two M tiles per wave

    // Per-thread fixed staging slots: 2 x b128 per matrix per tile.
    const int r0 = tid >> 3,           c0 = (tid & 7) << 3;
    const int r1 = (tid + 256) >> 3,   c1 = ((tid + 256) & 7) << 3;

    auto stage = [&](int buf, int k0) {
        async_copy_b128(&As[buf][r0][c0], &A[(size_t)(wgM + r0) * K + k0 + c0]);
        async_copy_b128(&Bs[buf][r0][c0], &Bm[(size_t)(wgN + r0) * K + k0 + c0]);
        async_copy_b128(&As[buf][r1][c1], &A[(size_t)(wgM + r1) * K + k0 + c1]);
        async_copy_b128(&Bs[buf][r1][c1], &Bm[(size_t)(wgN + r1) * K + k0 + c1]);
    };

    v8f acc0 = {}; v8f acc1 = {};
    const int nk = K >> 6;
    stage(0, 0);                               // prologue: tile 0
#pragma unroll 1
    for (int kt = 0; kt < nk; ++kt) {
        wait_async0();                         // tile kt landed (per-wave, in-order)
        __syncthreads();                       // all waves have it; prior buf reads done
        if (kt + 1 < nk) stage((kt + 1) & 1, (kt + 1) << 6);  // overlaps WMMAs below
        const int buf = kt & 1;
#pragma unroll
        for (int kk = 0; kk < 64; kk += 32) {
            const v16bf bf = load_fragB(&Bs[buf][ntl * 16][kk], 72, lane);
            const v16bf a0 = load_fragA(&As[buf][mBase][kk], 72, lane);
            const v16bf a1 = load_fragA(&As[buf][mBase + 16][kk], 72, lane);
            acc0 = __builtin_amdgcn_wmma_f32_16x16x32_bf16(false, a0, false, bf, (short)0, acc0, false, false);
            acc1 = __builtin_amdgcn_wmma_f32_16x16x32_bf16(false, a1, false, bf, (short)0, acc1, false, false);
        }
        __syncthreads();                       // reads done before buf is re-staged
    }

    const int nl = lane & 15, mh = (lane >> 4) * 8;
    const int col = wgN + ntl * 16 + nl;
    const float bv = bias ? bias[col] : 0.0f;
#pragma unroll
    for (int r = 0; r < 8; ++r) {
        const int m0 = wgM + mBase + mh + r;
        const int m1 = m0 + 16;
        const float v0 = acc0[r] + bv;
        const float v1 = acc1[r] + bv;
        if (Cf)  { Cf[(size_t)m0 * N + col] = v0;          Cf[(size_t)m1 * N + col] = v1; }
        if (Cbf) { Cbf[(size_t)m0 * N + col] = (__bf16)v0; Cbf[(size_t)m1 * N + col] = (__bf16)v1; }
    }
}

// ---------------------------------------------------------------------------
// Recurrent scan: h_{t+1} = tanh(xp[:,t,:] + h_t @ Whh^T), 512 steps.
// NWG=8 persistent WGs x 512 threads (16 waves). Each WG owns 64 output cols
// and keeps its Whh slice [64,512] bf16 in LDS for the whole kernel.
// h ping-pongs through two 64KB bf16 global buffers (L2-resident);
// device-scope atomic barrier between steps. h staging: 4 K-quarters,
// double-buffered async->LDS so copy(q+1) overlaps WMMAs(q).
// ---------------------------------------------------------------------------
__global__ __launch_bounds__(512) void rnn_scan(
    const float* __restrict__ xp, const __bf16* __restrict__ Whh,
    __bf16* __restrict__ hbuf0, __bf16* __restrict__ hbuf1,
    __bf16* __restrict__ outseq, unsigned int* __restrict__ bar) {
    __shared__ __bf16 Wsl[64][520];       // 66,560 B : resident weight slice
    __shared__ __bf16 Hs[2][64][136];     // 34,816 B : double-buffered h K-quarter
    const int wg   = blockIdx.x;
    const int tid  = threadIdx.x;
    const int lane = tid & 31, wave = tid >> 5;
    const int mt  = wave & 3;        // M tile (B rows)  0..3
    const int ntl = wave >> 2;       // local N tile     0..3

    // Preload weight slice (async): rows n_local in [0,64), global row wg*64+n_local.
#pragma unroll
    for (int j = 0; j < 8; ++j) {    // 64 rows * 64 chunks, 512 threads
        const int i = tid + j * 512;
        const int r = i >> 6, c = (i & 63) << 3;
        async_copy_b128(&Wsl[r][c], &Whh[(size_t)(wg * 64 + r) * H_ + c]);
    }

    // Per-thread fixed staging slots for one h K-quarter: 2 x b128.
    // Quarter = 64 rows x 128 cols bf16 = 1024 b128-chunks over 512 threads.
    const void* ldsQ[2][2];
    size_t goffQ[2];
#pragma unroll
    for (int j = 0; j < 2; ++j) {
        const int i = tid + j * 512;
        const int r = i >> 4, c = (i & 15) << 3;     // r in [0,64), c in [0,128)
        ldsQ[0][j] = &Hs[0][r][c];
        ldsQ[1][j] = &Hs[1][r][c];
        goffQ[j]   = (size_t)(r * H_ + c) * sizeof(__bf16);
    }

    const int nl = lane & 15, mh = (lane >> 4) * 8;
    const int col = wg * 64 + ntl * 16 + nl;

    for (int t = 0; t < T_; ++t) {
        const __bf16* hin  = (t & 1) ? hbuf1 : hbuf0;
        __bf16*       hout = (t & 1) ? hbuf0 : hbuf1;
        const char*   hinB = (const char*)hin;

        // Issue quarter 0 first: overlaps the xp f32 loads below.
        async_copy_b128(ldsQ[0][0], hinB + goffQ[0]);
        async_copy_b128(ldsQ[0][1], hinB + goffQ[1]);

        v8f acc;
#pragma unroll
        for (int r = 0; r < 8; ++r) {                     // init with xp (f32)
            const int b = mt * 16 + mh + r;
            acc[r] = xp[((size_t)b * T_ + t) * H_ + col]; // coalesced per 16-lane row
        }
        if (t + 1 < T_)  // warm L2 for next step's xp slice (global_prefetch_b8)
            __builtin_prefetch(&xp[((size_t)(mt * 16 + mh) * T_ + t + 1) * H_ + col], 0, 1);

#pragma unroll 1
        for (int q = 0; q < 4; ++q) {                     // four K-quarters of 128
            wait_async0();            // quarter q landed (first iter: + Wsl preload)
            __syncthreads();          // all waves ready; prior reads of other buf done
            if (q < 3) {              // stage q+1 while computing q
                const size_t go = (size_t)((q + 1) << 7) * sizeof(__bf16);
                async_copy_b128(ldsQ[(q + 1) & 1][0], hinB + go + goffQ[0]);
                async_copy_b128(ldsQ[(q + 1) & 1][1], hinB + go + goffQ[1]);
            }
            const int buf = q & 1, kq = q << 7;
#pragma unroll
            for (int kt = 0; kt < 128; kt += 32) {
                const v16bf a  = load_fragA(&Hs[buf][mt * 16][kt], 136, lane);
                const v16bf bf = load_fragB(&Wsl[ntl * 16][kq + kt], 520, lane);
                acc = __builtin_amdgcn_wmma_f32_16x16x32_bf16(false, a, false, bf, (short)0, acc, false, false);
            }
        }

#pragma unroll
        for (int r = 0; r < 8; ++r) {
            const int b = mt * 16 + mh + r;
            const float v = tanhf(acc[r]);
            const __bf16 hv = (__bf16)v;
            hout[b * H_ + col] = hv;
            if (outseq) outseq[((size_t)b * T_ + t) * H_ + col] = hv;
        }

        // ---- device-wide step barrier (release h_out, acquire for next read)
        __threadfence();
        __syncthreads();
        if (tid == 0) {
            __hip_atomic_fetch_add(bar, 1u, __ATOMIC_ACQ_REL, __HIP_MEMORY_SCOPE_AGENT);
            const unsigned int target = (unsigned)NWG * (unsigned)(t + 1);
            while (__hip_atomic_load(bar, __ATOMIC_ACQUIRE, __HIP_MEMORY_SCOPE_AGENT) < target)
                __builtin_amdgcn_s_sleep(1);
        }
        __syncthreads();
        __threadfence();
    }
}

// --------------------------- small helper kernels --------------------------
__global__ void f32_to_bf16_k(const float* __restrict__ s, __bf16* __restrict__ d, int n) {
    const int i = blockIdx.x * blockDim.x + threadIdx.x;
    if (i < n) d[i] = (__bf16)s[i];
}
__global__ void add_bias_k(const float* __restrict__ a, const float* __restrict__ b,
                           float* __restrict__ c, int n) {
    const int i = blockIdx.x * blockDim.x + threadIdx.x;
    if (i < n) c[i] = a[i] + b[i];
}
__global__ void scan_reset_k(__bf16* __restrict__ h0, unsigned int* __restrict__ bar) {
    const int i = blockIdx.x * blockDim.x + threadIdx.x;
    if (i < B_ * H_) h0[i] = (__bf16)0.0f;
    if (i == 0) *bar = 0u;
}

// ---------------------------------------------------------------------------
extern "C" void kernel_launch(void* const* d_in, const int* in_sizes, int n_in,
                              void* d_out, int out_size, void* d_ws, size_t ws_size,
                              hipStream_t stream) {
    const float* x     = (const float*)d_in[0];
    const float* W_ih0 = (const float*)d_in[1];
    const float* W_hh0 = (const float*)d_in[2];
    const float* b_ih0 = (const float*)d_in[3];
    const float* b_hh0 = (const float*)d_in[4];
    const float* W_ih1 = (const float*)d_in[5];
    const float* W_hh1 = (const float*)d_in[6];
    const float* b_ih1 = (const float*)d_in[7];
    const float* b_hh1 = (const float*)d_in[8];
    const float* W_out = (const float*)d_in[9];
    const float* b_out = (const float*)d_in[10];

    char* ws = (char*)d_ws;
    size_t off = 0;
    auto alloc = [&](size_t bytes) -> void* {
        void* p = ws + off;
        off += (bytes + 255) & ~(size_t)255;
        return p;
    };
    __bf16* x_bf     = (__bf16*)alloc((size_t)BT_ * IN_ * 2);   // 16 MB
    __bf16* wih0_bf  = (__bf16*)alloc((size_t)H_ * IN_ * 2);
    __bf16* whh0_bf  = (__bf16*)alloc((size_t)H_ * H_ * 2);
    __bf16* wih1_bf  = (__bf16*)alloc((size_t)H_ * H_ * 2);
    __bf16* whh1_bf  = (__bf16*)alloc((size_t)H_ * H_ * 2);
    __bf16* wout_bf  = (__bf16*)alloc((size_t)OUT_ * H_ * 2);
    float*  b0       = (float*) alloc((size_t)H_ * 4);
    float*  b1       = (float*) alloc((size_t)H_ * 4);
    float*  xp       = (float*) alloc((size_t)BT_ * H_ * 4);    // 64 MB (reused L0/L1)
    __bf16* out0_bf  = (__bf16*)alloc((size_t)BT_ * H_ * 2);    // 32 MB
    __bf16* hbuf0    = (__bf16*)alloc((size_t)B_ * H_ * 2);
    __bf16* hbuf1    = (__bf16*)alloc((size_t)B_ * H_ * 2);
    unsigned int* bar = (unsigned int*)alloc(256);
    (void)ws_size; (void)in_sizes; (void)n_in; (void)out_size;

    auto cvt = [&](const float* s, __bf16* d, int n) {
        f32_to_bf16_k<<<(n + 255) / 256, 256, 0, stream>>>(s, d, n);
    };
    cvt(x,     x_bf,    BT_ * IN_);
    cvt(W_ih0, wih0_bf, H_ * IN_);
    cvt(W_hh0, whh0_bf, H_ * H_);
    cvt(W_ih1, wih1_bf, H_ * H_);
    cvt(W_hh1, whh1_bf, H_ * H_);
    cvt(W_out, wout_bf, OUT_ * H_);
    add_bias_k<<<2, 256, 0, stream>>>(b_ih0, b_hh0, b0, H_);
    add_bias_k<<<2, 256, 0, stream>>>(b_ih1, b_hh1, b1, H_);

    // Layer 0 input projection: xp = x @ W_ih0^T + b0   [32768,512], K=256
    gemm_nt<<<dim3(H_ / 64, BT_ / 64), 256, 0, stream>>>(
        x_bf, wih0_bf, b0, xp, ((__bf16*)nullptr), BT_, H_, IN_);

    // Layer 0 recurrent scan (writes out0_bf)
    scan_reset_k<<<(B_ * H_ + 255) / 256, 256, 0, stream>>>(hbuf0, bar);
    rnn_scan<<<NWG, 512, 0, stream>>>(xp, whh0_bf, hbuf0, hbuf1, out0_bf, bar);

    // Layer 1 input projection: xp = out0 @ W_ih1^T + b1   [32768,512], K=512
    gemm_nt<<<dim3(H_ / 64, BT_ / 64), 256, 0, stream>>>(
        out0_bf, wih1_bf, b1, xp, ((__bf16*)nullptr), BT_, H_, H_);

    // Layer 1 recurrent scan (only h_last needed; T even -> lands in hbuf0)
    scan_reset_k<<<(B_ * H_ + 255) / 256, 256, 0, stream>>>(hbuf0, bar);
    rnn_scan<<<NWG, 512, 0, stream>>>(xp, whh1_bf, hbuf0, hbuf1, (__bf16*)nullptr, bar);

    // Output: d_out[64,256] = h_last @ W_out^T + b_out
    gemm_nt<<<dim3(OUT_ / 64, B_ / 64), 256, 0, stream>>>(
        hbuf0, wout_bf, b_out, (float*)d_out, (__bf16*)nullptr, B_, OUT_, H_);
}
